// _SmoothLinear_23802708754444
// MI455X (gfx1250) — matrix operand order repaired and checked by reading, not compiled, not run on hardware
//
#include <hip/hip_runtime.h>
#include <hip/hip_bf16.h>

typedef __attribute__((ext_vector_type(16))) _Float16 v16h;
typedef __attribute__((ext_vector_type(8)))  _Float16 v8h;
typedef __attribute__((ext_vector_type(8)))  float    v8f;
typedef __attribute__((ext_vector_type(4)))  float    v4f;

#define OUT_F 4096
#define IN_F  4096
#define M_TOT 8192   // 4 * 2048
#define KDIM  4096

// ---------------------------------------------------------------------------
// CDNA5 async global->LDS copy (b128), GV addressing.  The LDS destination is
// carried in a VGPR as a 32-bit LDS offset (low 32 bits of the flat shared
// pointer).  Tracked with ASYNCcnt.
// ---------------------------------------------------------------------------
__device__ __forceinline__ void async_copy_b128(unsigned lds_off, const void* gaddr)
{
    asm volatile("global_load_async_to_lds_b128 %0, %1, off"
                 :: "v"(lds_off), "v"(gaddr)
                 : "memory");
}
__device__ __forceinline__ void wait_async_le4()
{
    asm volatile("s_wait_asynccnt 0x4" ::: "memory");
}
__device__ __forceinline__ void wait_async_0()
{
    asm volatile("s_wait_asynccnt 0x0" ::: "memory");
}

// ---------------------------------------------------------------------------
// Kernel 1: VQ decompression.  One thread per 8-element codebook block.
// Folds scale[i] into the weight (scale indexes the K/IN dimension), converts
// to f16, stores one packed v8h (global_store_b128).
// ---------------------------------------------------------------------------
__global__ __launch_bounds__(256)
void vq_decompress_w(const float* __restrict__ centroids,   // [256][8]
                     const long long* __restrict__ labels,  // [OUT*IN/8] int64
                     const float* __restrict__ scale,       // [IN]
                     _Float16* __restrict__ Wh)             // [OUT][IN] f16
{
    int t = blockIdx.x * blockDim.x + threadIdx.x;          // 0 .. 2M-1
    long long lab = labels[t];
    const float* c = centroids + ((int)lab << 3);
    int i0 = (t << 3) & (IN_F - 1);
    v8h w;
#pragma unroll
    for (int j = 0; j < 8; ++j)
        w[j] = (_Float16)(c[j] * scale[i0 + j]);
    *(v8h*)(Wh + ((size_t)t << 3)) = w;
}

// ---------------------------------------------------------------------------
// Kernel 2: x fp32 -> f16 (8 elements / thread).  x is read exactly once ->
// nontemporal loads to keep it out of L2 (which we want holding W / xh).
// ---------------------------------------------------------------------------
__global__ __launch_bounds__(256)
void convert_x_f16(const float* __restrict__ x, _Float16* __restrict__ xh)
{
    size_t t = ((size_t)blockIdx.x * blockDim.x + threadIdx.x) << 3;
    v4f a = __builtin_nontemporal_load((const v4f*)(x + t));
    v4f b = __builtin_nontemporal_load((const v4f*)(x + t + 4));
    v8h h;
#pragma unroll
    for (int j = 0; j < 4; ++j) { h[j] = (_Float16)a[j]; h[j + 4] = (_Float16)b[j]; }
    *(v8h*)(xh + t) = h;
}

__device__ __forceinline__ v16h cat16(v8h lo, v8h hi)
{
    v16h r;
#pragma unroll
    for (int i = 0; i < 8; ++i) { r[i] = lo[i]; r[i + 8] = hi[i]; }
    return r;
}

// ---------------------------------------------------------------------------
// Kernel 3: WMMA GEMM.  out[m,n] = sum_k xh[m,k]*Wh[n,k] + bias[n]
// Block tile 128x128, K-step 32, 8 wave32s (2x4), each wave: 64x32 tile
// = 4x2 accumulators.  Double-buffered LDS fed by async global->LDS DMA.
// ---------------------------------------------------------------------------
__global__ __launch_bounds__(256)
void vq_gemm_wmma(const _Float16* __restrict__ A,   // xh  [8192][4096]
                  const _Float16* __restrict__ B,   // Wh  [4096][4096] (row o, col k)
                  const float* __restrict__ bias,   // [4096]
                  float* __restrict__ C)            // [8192][4096]
{
    constexpr int BM = 128, BN = 128, BK = 32;
    constexpr int LST = BK + 8;                     // padded LDS row stride (halves)
    constexpr int NSTEP = KDIM / BK;                // 128

    __shared__ _Float16 As[2][BM * LST];
    __shared__ _Float16 Bs[2][BN * LST];

    const int tid  = threadIdx.x;
    const int lane = tid & 31;
    const int wave = tid >> 5;       // 0..7
    const int wm   = wave >> 2;      // 0..1  (M direction)
    const int wn   = wave & 3;       // 0..3  (N direction)
    const int hl   = lane >> 4;      // lane half (0/1)
    const int r    = lane & 15;

    const int m0 = blockIdx.y * BM;
    const int n0 = blockIdx.x * BN;

    // Per-thread staging slots: idx in [0,512): row = idx>>2, chunk = idx&3
    const int row0 = tid >> 2;                 // rows 0..63
    const int row1 = (tid + 256) >> 2;         // rows 64..127
    const int ch0  = (tid & 3) << 3;           // 0,8,16,24 halves
    const _Float16* Arow0 = A + (size_t)(m0 + row0) * KDIM + ch0;
    const _Float16* Arow1 = A + (size_t)(m0 + row1) * KDIM + ch0;
    const _Float16* Brow0 = B + (size_t)(n0 + row0) * KDIM + ch0;
    const _Float16* Brow1 = B + (size_t)(n0 + row1) * KDIM + ch0;

    v8f acc[4][2];
#pragma unroll
    for (int mt = 0; mt < 4; ++mt)
#pragma unroll
        for (int nt = 0; nt < 2; ++nt)
#pragma unroll
            for (int g = 0; g < 8; ++g) acc[mt][nt][g] = 0.0f;

    // -------- prologue: DMA stage 0 into buffer 0 --------
    {
        unsigned a0 = (unsigned)(uintptr_t)&As[0][row0 * LST + ch0];
        unsigned a1 = (unsigned)(uintptr_t)&As[0][row1 * LST + ch0];
        unsigned b0 = (unsigned)(uintptr_t)&Bs[0][row0 * LST + ch0];
        unsigned b1 = (unsigned)(uintptr_t)&Bs[0][row1 * LST + ch0];
        async_copy_b128(a0, Arow0);
        async_copy_b128(a1, Arow1);
        async_copy_b128(b0, Brow0);
        async_copy_b128(b1, Brow1);
    }

    for (int i = 0; i < NSTEP; ++i) {
        const int buf = i & 1;
        // -------- issue DMA for stage i+1 into the other buffer --------
        if (i + 1 < NSTEP) {
            const int kb1 = (i + 1) * BK;
            const int nb  = buf ^ 1;
            unsigned a0 = (unsigned)(uintptr_t)&As[nb][row0 * LST + ch0];
            unsigned a1 = (unsigned)(uintptr_t)&As[nb][row1 * LST + ch0];
            unsigned b0 = (unsigned)(uintptr_t)&Bs[nb][row0 * LST + ch0];
            unsigned b1 = (unsigned)(uintptr_t)&Bs[nb][row1 * LST + ch0];
            async_copy_b128(a0, Arow0 + kb1);
            async_copy_b128(a1, Arow1 + kb1);
            async_copy_b128(b0, Brow0 + kb1);
            async_copy_b128(b1, Brow1 + kb1);
            wait_async_le4();      // stage i's 4 copies done; stage i+1 in flight
        } else {
            wait_async_0();        // last stage: drain everything
        }
        __syncthreads();           // all waves' stage-i data visible in LDS

        // -------- fragments per ISA VGPR layouts --------
        // A 16x32 f16: lane half 0 -> K {0..7,16..23}; half 1 -> K {8..15,24..31}
        v16h af[4];
#pragma unroll
        for (int mt = 0; mt < 4; ++mt) {
            int m = wm * 64 + mt * 16 + r;
            v8h lo = *(const v8h*)(&As[buf][m * LST + hl * 8]);
            v8h hi = *(const v8h*)(&As[buf][m * LST + 16 + hl * 8]);
            af[mt] = cat16(lo, hi);
        }
        // B 32x16 f16: lane half 0 -> K {0..7,16..23}; half 1 -> K {8..15,24..31}; N = lane&15
        v16h bf[2];
#pragma unroll
        for (int nt = 0; nt < 2; ++nt) {
            int n = wn * 32 + nt * 16 + r;
            v8h lo = *(const v8h*)(&Bs[buf][n * LST + hl * 8]);
            v8h hi = *(const v8h*)(&Bs[buf][n * LST + 16 + hl * 8]);
            bf[nt] = cat16(lo, hi);
        }

        // -------- 8 WMMAs per K-step per wave --------
#pragma unroll
        for (int mt = 0; mt < 4; ++mt)
#pragma unroll
            for (int nt = 0; nt < 2; ++nt)
                acc[mt][nt] = __builtin_amdgcn_wmma_f32_16x16x32_f16(
                    false, af[mt], false, bf[nt],
                    (short)0, acc[mt][nt], false, false);

        __syncthreads();           // all waves done reading buf before stage i+2 lands
    }

    // -------- epilogue: C/D layout -> row = half*8 + g, col = lane&15 --------
#pragma unroll
    for (int nt = 0; nt < 2; ++nt) {
        int n = n0 + wn * 32 + nt * 16 + r;
        float bv = bias[n];
#pragma unroll
        for (int mt = 0; mt < 4; ++mt) {
#pragma unroll
            for (int g = 0; g < 8; ++g) {
                int m = m0 + wm * 64 + mt * 16 + hl * 8 + g;
                __builtin_nontemporal_store(acc[mt][nt][g] + bv,
                                            C + (size_t)m * OUT_F + n);
            }
        }
    }
}

// ---------------------------------------------------------------------------
// Host-side launcher.  d_in order: x, centroids, labels, scale, bias.
// Workspace: xh (8192*4096 f16 = 64 MiB) followed by Wh (4096*4096 f16 = 32 MiB).
// ---------------------------------------------------------------------------
extern "C" void kernel_launch(void* const* d_in, const int* in_sizes, int n_in,
                              void* d_out, int out_size, void* d_ws, size_t ws_size,
                              hipStream_t stream)
{
    const float*     x         = (const float*)d_in[0];
    const float*     centroids = (const float*)d_in[1];
    const long long* labels    = (const long long*)d_in[2];
    const float*     scale     = (const float*)d_in[3];
    const float*     bias      = (const float*)d_in[4];
    float*           out       = (float*)d_out;

    _Float16* xh = (_Float16*)d_ws;
    _Float16* Wh = (_Float16*)((char*)d_ws + (size_t)M_TOT * KDIM * sizeof(_Float16));

    // 1) decompress + scale-fold weights: 2M blocks of 8
    vq_decompress_w<<<(OUT_F * IN_F / 8) / 256, 256, 0, stream>>>(centroids, labels, scale, Wh);

    // 2) x -> f16: 33.5M elements, 8 per thread
    convert_x_f16<<<(M_TOT * KDIM / 8) / 256, 256, 0, stream>>>(x, xh);

    // 3) WMMA GEMM: grid (N/128, M/128) = (32, 64)
    dim3 grid(OUT_F / 128, M_TOT / 128);
    vq_gemm_wmma<<<grid, 256, 0, stream>>>(xh, Wh, bias, out);
}
